// JanossyPooling_7842610282597
// MI455X (gfx1250) — matrix operand order repaired and checked by reading, not compiled, not run on hardware
//
#include <hip/hip_runtime.h>
#include <hip/hip_bf16.h>

typedef __attribute__((ext_vector_type(16))) _Float16 v16h;
typedef __attribute__((ext_vector_type(8)))  _Float16 v8h;
typedef __attribute__((ext_vector_type(4)))  _Float16 v4h;
typedef __attribute__((ext_vector_type(8)))  float    v8f;

// ---------------------------------------------------------------------------
// Weight conversion: W [K][N] fp32 row-major  ->  WT [N][K] f16 row-major.
// ---------------------------------------------------------------------------
__global__ __launch_bounds__(256) void convert_transpose_kernel(
    const float* __restrict__ W, _Float16* __restrict__ WT, int K, int N) {
  int i = blockIdx.x * 256 + threadIdx.x;
  if (i < K * N) {
    int k = i / N;
    int n = i - k * N;
    WT[(size_t)n * K + k] = (_Float16)W[i];
  }
}

// ---------------------------------------------------------------------------
// One dense layer on a 32-row tile: y[32 x 512] = relu(x[32 x K] @ W + b).
// 8 wave32's; each wave owns 4 N-tiles x 2 M-tiles with a K-outer loop:
// per K-tile the two A fragments are loaded ONCE from LDS and reused by all
// 4 N-tiles (8 WMMAs), and each B fragment feeds 2 WMMAs. 8 independent
// accumulator chains -> no WMMA RAW hazards.
// ---------------------------------------------------------------------------
template <int K>
__device__ __forceinline__ void dense_layer_wmma(
    const _Float16* __restrict__ x, int xstride,
    _Float16* __restrict__ y, int ystride,
    const _Float16* __restrict__ wt, const float* __restrict__ bias,
    bool do_relu) {
  const int lane = threadIdx.x & 31;
  const int wave = threadIdx.x >> 5;
  const int mlow = lane & 15;
  const int hi   = (lane & 16) ? 1 : 0;

  // A fragment bases for the two M-tiles (rows mlow and 16+mlow).
  const _Float16* xrow0 = x + mlow * xstride + (hi ? 8 : 0);
  const _Float16* xrow1 = xrow0 + 16 * xstride;

  // B fragment base for N-tile 0 of this wave; tiles are 16*K halves apart.
  const int ncol0 = wave * 64 + mlow;
  const _Float16* wrow = wt + (size_t)ncol0 * K + (hi ? 8 : 0);

  v8f acc[4][2];
#pragma unroll
  for (int t = 0; t < 4; ++t) {
    acc[t][0] = (v8f){};
    acc[t][1] = (v8f){};
  }

  for (int k0 = 0; k0 < K; k0 += 32) {
    // A: 16-bit 16x32 layout -> two 16B LDS chunks per M-tile, loaded once.
    v8h a0lo = *(const v8h*)(xrow0 + k0);
    v8h a0hi = *(const v8h*)(xrow0 + k0 + 16);
    v8h a1lo = *(const v8h*)(xrow1 + k0);
    v8h a1hi = *(const v8h*)(xrow1 + k0 + 16);
    v16h a0, a1;
#pragma unroll
    for (int i = 0; i < 8; ++i) {
      a0[i] = a0lo[i]; a0[8 + i] = a0hi[i];
      a1[i] = a1lo[i]; a1[8 + i] = a1hi[i];
    }
#pragma unroll
    for (int t = 0; t < 4; ++t) {
      const _Float16* wp = wrow + (size_t)t * 16 * K + k0;
      v8h blo = *(const v8h*)(wp);
      v8h bhi = *(const v8h*)(wp + 16);
      v16h b;
#pragma unroll
      for (int i = 0; i < 8; ++i) { b[i] = blo[i]; b[8 + i] = bhi[i]; }
      acc[t][0] = __builtin_amdgcn_wmma_f32_16x16x32_f16(
          false, a0, false, b, (short)0, acc[t][0], false, false);
      acc[t][1] = __builtin_amdgcn_wmma_f32_16x16x32_f16(
          false, a1, false, b, (short)0, acc[t][1], false, false);
    }
  }

  const int mbase = hi ? 8 : 0;
#pragma unroll
  for (int t = 0; t < 4; ++t) {
    const int ncol = ncol0 + t * 16;
    const float bv = bias[ncol];
#pragma unroll
    for (int r = 0; r < 8; ++r) {
      float v0 = acc[t][0][r] + bv;
      float v1 = acc[t][1][r] + bv;
      if (do_relu) {
        v0 = v0 > 0.f ? v0 : 0.f;
        v1 = v1 > 0.f ? v1 : 0.f;
      }
      y[(mbase + r) * ystride + ncol]      = (_Float16)v0;
      y[(16 + mbase + r) * ystride + ncol] = (_Float16)v1;
    }
  }
}

// ---------------------------------------------------------------------------
// Fused Janossy-level MLP. One block = one 32-row tile.
//   L==1 : x = h[row]                            (DIN = 256)
//   L>=2 : x block j = h[idx[j]] + h[idx[L-1-j]] (DIN = 256*L)
// Layers: DIN->512 relu, 512->512 relu, 512->512 relu, 512->2.
// ---------------------------------------------------------------------------
template <int L, int DIN>
__global__ __launch_bounds__(256) void janossy_mlp_kernel(
    const float* __restrict__ h, const int* __restrict__ idx, int nrows,
    const _Float16* __restrict__ wt0, const _Float16* __restrict__ wt1,
    const _Float16* __restrict__ wt2, const _Float16* __restrict__ wot,
    const float* __restrict__ b0, const float* __restrict__ b1,
    const float* __restrict__ b2, const float* __restrict__ bo,
    float* __restrict__ out) {
  constexpr int XW = ((DIN < 512) ? 512 : DIN) + 8;  // padded stride (halves)
  constexpr int YW = 512 + 8;

  __shared__ __align__(16) _Float16 xbuf[32 * XW];
  __shared__ __align__(16) _Float16 ybuf[32 * YW];
  __shared__ float osum[64 * 4];

  const int row0 = blockIdx.x * 32;
  const int tid  = threadIdx.x;

  // ---- Stage 0: build symmetrized input tile in LDS (f16), 4-wide ----
  for (int e = tid * 4; e < 32 * DIN; e += 1024) {
    const int r   = e / DIN;
    const int c   = e - r * DIN;
    const int row = row0 + r;
    float4 v = make_float4(0.f, 0.f, 0.f, 0.f);
    if (row < nrows) {
      if (L == 1) {
        v = *(const float4*)(h + (size_t)row * 256 + c);
      } else {
        const int j  = c >> 8;      // 256-wide block
        const int cc = c & 255;
        const int ia = idx[row * L + j];
        const int ib = idx[row * L + (L - 1 - j)];
        const float4 va = *(const float4*)(h + (size_t)ia * 256 + cc);
        const float4 vb = *(const float4*)(h + (size_t)ib * 256 + cc);
        v = make_float4(va.x + vb.x, va.y + vb.y, va.z + vb.z, va.w + vb.w);
      }
    }
    v4h hv;
    hv[0] = (_Float16)v.x; hv[1] = (_Float16)v.y;
    hv[2] = (_Float16)v.z; hv[3] = (_Float16)v.w;
    *(v4h*)(xbuf + r * XW + c) = hv;
  }
  __syncthreads();

  // ---- Hidden layers on matrix cores ----
  dense_layer_wmma<DIN>(xbuf, XW, ybuf, YW, wt0, b0, true);
  __syncthreads();
  dense_layer_wmma<512>(ybuf, YW, xbuf, XW, wt1, b1, true);
  __syncthreads();
  dense_layer_wmma<512>(xbuf, XW, ybuf, YW, wt2, b2, true);
  __syncthreads();

  // ---- Output layer 512 -> 2, parallel over all 256 threads ----
  {
    const int m  = tid & 31;         // row within tile
    const int c  = (tid >> 5) & 1;   // output column
    const int kc = tid >> 6;         // K-chunk 0..3 (128 each)
    const _Float16* xr = ybuf + m * YW + kc * 128;
    const _Float16* wr = wot + c * 512 + kc * 128;
    float partial = 0.f;
#pragma unroll
    for (int k8 = 0; k8 < 16; ++k8) {
      v8h xv = *(const v8h*)(xr + k8 * 8);
      v8h wv = *(const v8h*)(wr + k8 * 8);
#pragma unroll
      for (int i = 0; i < 8; ++i) partial += (float)xv[i] * (float)wv[i];
    }
    osum[(m * 2 + c) * 4 + kc] = partial;
  }
  __syncthreads();
  if (tid < 64) {
    const int m   = tid >> 1;
    const int c   = tid & 1;
    const int row = row0 + m;
    if (row < nrows) {
      float s = bo[c] + osum[tid * 4 + 0] + osum[tid * 4 + 1] +
                osum[tid * 4 + 2] + osum[tid * 4 + 3];
      out[(size_t)row * 2 + c] = s;
    }
  }
}

// ---------------------------------------------------------------------------
// Host side
// ---------------------------------------------------------------------------
struct MlpParams {
  const float *W0, *W1, *W2, *Wo, *b0, *b1, *b2, *bo;
};

static inline void launch_convert(const float* W, _Float16* WT, int K, int N,
                                  hipStream_t s) {
  int total = K * N;
  convert_transpose_kernel<<<(total + 255) / 256, 256, 0, s>>>(W, WT, K, N);
}

extern "C" void kernel_launch(void* const* d_in, const int* in_sizes, int n_in,
                              void* d_out, int out_size, void* d_ws,
                              size_t ws_size, hipStream_t stream) {
  // Inputs (JAX pytree flatten, sorted keys):
  // 0:h 1:idx2 2:idx3 3:idx4, then per pN: W0,W1,W2,Wo,b0,b1,b2,bo
  const float* h    = (const float*)d_in[0];
  const int*   idx2 = (const int*)d_in[1];
  const int*   idx3 = (const int*)d_in[2];
  const int*   idx4 = (const int*)d_in[3];

  MlpParams p[4];
  for (int i = 0; i < 4; ++i) {
    const int base = 4 + i * 8;
    p[i].W0 = (const float*)d_in[base + 0];
    p[i].W1 = (const float*)d_in[base + 1];
    p[i].W2 = (const float*)d_in[base + 2];
    p[i].Wo = (const float*)d_in[base + 3];
    p[i].b0 = (const float*)d_in[base + 4];
    p[i].b1 = (const float*)d_in[base + 5];
    p[i].b2 = (const float*)d_in[base + 6];
    p[i].bo = (const float*)d_in[base + 7];
  }

  const int n1 = in_sizes[0] / 256;  // 100000
  const int n2 = in_sizes[1] / 2;    // 100000
  const int n3 = in_sizes[2] / 3;    // 200000
  const int n4 = in_sizes[3] / 4;    // 300000

  // Workspace: f16 transposed weights per level.
  _Float16* ws  = (_Float16*)d_ws;
  size_t    off = 0;
  _Float16 *wt0[4], *wt1[4], *wt2[4], *wot[4];
  const int din[4] = {256, 512, 768, 1024};
  for (int i = 0; i < 4; ++i) {
    wt0[i] = ws + off; off += (size_t)512 * din[i];
    wt1[i] = ws + off; off += (size_t)512 * 512;
    wt2[i] = ws + off; off += (size_t)512 * 512;
    wot[i] = ws + off; off += (size_t)2 * 512;
  }
  for (int i = 0; i < 4; ++i) {
    launch_convert(p[i].W0, wt0[i], din[i], 512, stream);
    launch_convert(p[i].W1, wt1[i], 512, 512, stream);
    launch_convert(p[i].W2, wt2[i], 512, 512, stream);
    launch_convert(p[i].Wo, wot[i], 512, 2, stream);
  }

  float* out1 = (float*)d_out;
  float* out2 = out1 + (size_t)2 * n1;
  float* out3 = out2 + (size_t)2 * n2;
  float* out4 = out3 + (size_t)2 * n3;

  janossy_mlp_kernel<1, 256><<<(n1 + 31) / 32, 256, 0, stream>>>(
      h, nullptr, n1, wt0[0], wt1[0], wt2[0], wot[0],
      p[0].b0, p[0].b1, p[0].b2, p[0].bo, out1);
  janossy_mlp_kernel<2, 512><<<(n2 + 31) / 32, 256, 0, stream>>>(
      h, idx2, n2, wt0[1], wt1[1], wt2[1], wot[1],
      p[1].b0, p[1].b1, p[1].b2, p[1].bo, out2);
  janossy_mlp_kernel<3, 768><<<(n3 + 31) / 32, 256, 0, stream>>>(
      h, idx3, n3, wt0[2], wt1[2], wt2[2], wot[2],
      p[2].b0, p[2].b1, p[2].b2, p[2].bo, out3);
  janossy_mlp_kernel<4, 1024><<<(n4 + 31) / 32, 256, 0, stream>>>(
      h, idx4, n4, wt0[3], wt1[3], wt2[3], wot[3],
      p[3].b0, p[3].b1, p[3].b2, p[3].bo, out4);
}
